// Attention_661424963964
// MI455X (gfx1250) — matrix operand order repaired and checked by reading, not compiled, not run on hardware
//
#include <hip/hip_runtime.h>

#define SEQ_LEN   2048
#define D_MODEL_C 768
#define D_HEAD_C  64
#define N_HEADS_C 12
#define BATCH_C   4

typedef __attribute__((ext_vector_type(16))) __bf16 v16bf;
typedef __attribute__((ext_vector_type(8)))  float  v8f;
typedef __attribute__((ext_vector_type(16))) unsigned short v16us;

union Frag {
  v16us us;
  v16bf bf;
  uint4 q[2];
};

__device__ __forceinline__ unsigned short f2bf(float f) {
  // native bf16 convert (RNE) -> v_cvt_pk_bf16_f32 when pairable
  return __builtin_bit_cast(unsigned short, (__bf16)f);
}

__device__ __forceinline__ v8f wmma_bf16(v16bf a, v16bf b, v8f c) {
  // 8 args: (neg_a, A, neg_b, B, c_mod, C, reuse_a, reuse_b)
  return __builtin_amdgcn_wmma_f32_16x16x32_bf16(false, a, false, b, (short)0, c,
                                                 false, false);
}

// ---------------------------------------------------------------------------
// Kernel 1: per-head QKV projection.  X[b] (128-row tile x 768) @ W[h] (768x64).
// mode: 0 = Q (scale 1/8, row-major bf16 out [b,h,s,d])
//       1 = K (row-major bf16 out [b,h,s,d])
//       2 = V (transposed bf16 out [b,h,d,s])
// ---------------------------------------------------------------------------
__global__ __launch_bounds__(256) void qkv_kernel(
    const float* __restrict__ x, const float* __restrict__ W,
    const float* __restrict__ bias, unsigned short* __restrict__ out, int mode) {
  __shared__ unsigned short Wlds[D_HEAD_C * 32];  // [d=64][k=32] bf16, N-major

  const int tid  = threadIdx.x;
  const int w    = tid >> 5;
  const int lane = tid & 31;
  const int lnm  = lane & 15;
  const int lhi  = lane >> 4;
  const int h    = blockIdx.y;
  const int bt   = blockIdx.x;
  const int b    = bt / (SEQ_LEN / 128);
  const int s0   = (bt % (SEQ_LEN / 128)) * 128;

  const int kbase8  = lhi ? 8 : 0;    // A-fragment K split (16-bit A layout)
  const int kbase16 = lhi ? 16 : 0;   // unused; B fragments use kbase8 as well

  const int srow = s0 + 16 * w + lnm;                 // this lane's A row
  const float* xrow = x + (size_t)(b * SEQ_LEN + srow) * D_MODEL_C;

  v8f acc[4] = {v8f{}, v8f{}, v8f{}, v8f{}};

  for (int k0 = 0; k0 < D_MODEL_C; k0 += 32) {
    // Stage W[h][k0..k0+31][0..63] transposed (N-major) into LDS as bf16.
    {
      const int ml = tid & 31;
      const int d0 = (tid >> 5) * 8;
      const float* wr = W + ((size_t)h * D_MODEL_C + (k0 + ml)) * D_HEAD_C + d0;
      float4 w0 = *(const float4*)wr;
      float4 w1 = *(const float4*)(wr + 4);
      Wlds[(d0 + 0) * 32 + ml] = f2bf(w0.x);
      Wlds[(d0 + 1) * 32 + ml] = f2bf(w0.y);
      Wlds[(d0 + 2) * 32 + ml] = f2bf(w0.z);
      Wlds[(d0 + 3) * 32 + ml] = f2bf(w0.w);
      Wlds[(d0 + 4) * 32 + ml] = f2bf(w1.x);
      Wlds[(d0 + 5) * 32 + ml] = f2bf(w1.y);
      Wlds[(d0 + 6) * 32 + ml] = f2bf(w1.z);
      Wlds[(d0 + 7) * 32 + ml] = f2bf(w1.w);
    }
    __syncthreads();

    // A fragment: 16x32 bf16 of X (converted on the fly from f32).
    Frag a;
    {
      const float* p0 = xrow + k0 + kbase8;
      const float* p1 = xrow + k0 + kbase8 + 16;
#pragma unroll
      for (int i = 0; i < 8; ++i) a.us[i] = f2bf(p0[i]);
#pragma unroll
      for (int i = 0; i < 8; ++i) a.us[8 + i] = f2bf(p1[i]);
    }

    // Batch all B fragments, then issue the WMMA chain.
    Frag bf[4];
#pragma unroll
    for (int t = 0; t < 4; ++t) {
      const unsigned short* brow = Wlds + (16 * t + lnm) * 32 + kbase8;
      bf[t].q[0] = *(const uint4*)(brow);
      bf[t].q[1] = *(const uint4*)(brow + 16);
    }
#pragma unroll
    for (int t = 0; t < 4; ++t) acc[t] = wmma_bf16(a.bf, bf[t].bf, acc[t]);
    __syncthreads();
  }

  const int bh = b * N_HEADS_C + h;
  if (mode == 2) {
    // V: store transposed [b,h,d,s]; per lane the 8 acc rows are contiguous keys.
#pragma unroll
    for (int t = 0; t < 4; ++t) {
      const float bv = bias[h * D_HEAD_C + 16 * t + lnm];
      unsigned int pk[4];
#pragma unroll
      for (int r = 0; r < 4; ++r) {
        const float v0 = acc[t][2 * r] + bv;
        const float v1 = acc[t][2 * r + 1] + bv;
        pk[r] = (unsigned int)f2bf(v0) | ((unsigned int)f2bf(v1) << 16);
      }
      const int d    = 16 * t + lnm;
      const int key0 = s0 + 16 * w + 8 * lhi;
      *(uint4*)(out + (size_t)(bh * D_HEAD_C + d) * SEQ_LEN + key0) =
          *(uint4*)pk;
    }
  } else {
    const float scale = (mode == 0) ? 0.125f : 1.0f;  // fold 1/sqrt(64) into Q
#pragma unroll
    for (int t = 0; t < 4; ++t) {
      const float bv = bias[h * D_HEAD_C + 16 * t + lnm];
#pragma unroll
      for (int r = 0; r < 8; ++r) {
        const int q = s0 + 16 * w + r + 8 * lhi;  // C layout: M = r + 8*hi
        out[(size_t)(bh * SEQ_LEN + q) * D_HEAD_C + 16 * t + lnm] =
            f2bf((acc[t][r] + bv) * scale);
      }
    }
  }
}

// ---------------------------------------------------------------------------
// Async staging: each wave issues 4 GLOBAL_LOAD_ASYNC_TO_LDS_B128 (ASYNCcnt),
// together covering one 64x64 bf16 K tile and one 64x64 bf16 V tile.
// ---------------------------------------------------------------------------
__device__ __forceinline__ void stage_kv(unsigned ldsK, unsigned ldsV,
                                         unsigned long long kbase_g,
                                         unsigned long long vbase_g, int kb,
                                         int w, int srow, int scol) {
#pragma unroll
  for (int i = 0; i < 2; ++i) {
    const int row = 4 * (w + 8 * i) + srow;  // 0..63 across the block
    const unsigned kv = (unsigned)(((kb + row) * D_HEAD_C + scol) * 2);
    const unsigned kl = ldsK + (unsigned)((row * 64 + scol) * 2);
    asm volatile("global_load_async_to_lds_b128 %0, %1, %2"
                 :: "v"(kl), "v"(kv), "s"(kbase_g) : "memory");
    const unsigned vv = (unsigned)((row * SEQ_LEN + kb + scol) * 2);
    const unsigned vl = ldsV + (unsigned)((row * 64 + scol) * 2);
    asm volatile("global_load_async_to_lds_b128 %0, %1, %2"
                 :: "v"(vl), "v"(vv), "s"(vbase_g) : "memory");
  }
}

// ---------------------------------------------------------------------------
// Kernel 2: causal flash attention.  Double-buffered async K/V staging: the
// next 64-key chunk streams into LDS while the current one feeds the WMMAs.
// ---------------------------------------------------------------------------
__global__ __launch_bounds__(256) void attn_kernel(
    const unsigned short* __restrict__ Qb, const unsigned short* __restrict__ Kb,
    const unsigned short* __restrict__ Vt, unsigned short* __restrict__ Z) {
  __shared__ unsigned short Klds[2][64 * 64];   // [buf][key][d]
  __shared__ unsigned short Vlds[2][64 * 64];   // [buf][d][key]
  __shared__ unsigned short Plds[8 * 16 * 64];  // per-wave 16x64 bf16 P tile

  const int tid  = threadIdx.x;
  const int w    = tid >> 5;
  const int lane = tid & 31;
  const int lnm  = lane & 15;
  const int lhi  = lane >> 4;
  const int h    = blockIdx.y;
  const int b    = blockIdx.z;
  const int bh   = b * N_HEADS_C + h;
  const int q0   = blockIdx.x * 128;
  const int q0w  = q0 + 16 * w;                 // first q-row of this wave
  const int kbase8  = lhi ? 8 : 0;
  const int kbase16 = lhi ? 16 : 0;
  unsigned short* P = Plds + w * (16 * 64);

  const int srow = lane >> 3;        // 0..3   (row within 4-row group)
  const int scol = (lane & 7) * 8;   // 0..56  (element column, 16B granules)
  const unsigned ldsK = (unsigned)(uintptr_t)&Klds[0][0];
  const unsigned ldsV = (unsigned)(uintptr_t)&Vlds[0][0];
  const unsigned long long kbase_g =
      (unsigned long long)(uintptr_t)(Kb + (size_t)bh * SEQ_LEN * D_HEAD_C);
  const unsigned long long vbase_g =
      (unsigned long long)(uintptr_t)(Vt + (size_t)bh * D_HEAD_C * SEQ_LEN);

  // Q fragments (d_head = 64 -> two K=32 fragments), loaded once.
  Frag qa0, qa1;
  {
    const unsigned short* qrow =
        Qb + (size_t)(bh * SEQ_LEN + q0w + lnm) * D_HEAD_C;
    qa0.q[0] = *(const uint4*)(qrow + kbase8);
    qa0.q[1] = *(const uint4*)(qrow + kbase8 + 16);
    qa1.q[0] = *(const uint4*)(qrow + 32 + kbase8);
    qa1.q[1] = *(const uint4*)(qrow + 32 + kbase8 + 16);
  }

  float mrun[8], lrun[8];
#pragma unroll
  for (int r = 0; r < 8; ++r) { mrun[r] = -1e30f; lrun[r] = 0.0f; }
  v8f acc[4] = {v8f{}, v8f{}, v8f{}, v8f{}};

  const int nkeys_blk = q0 + 128;  // causal bound for the whole block

  // Prologue: stage chunk 0 into buffer 0.
  stage_kv(ldsK, ldsV, kbase_g, vbase_g, 0, w, srow, scol);
  asm volatile("s_wait_asynccnt 0x0" ::: "memory");
  __syncthreads();

  for (int kb = 0; kb < nkeys_blk; kb += 64) {
    const int cur = (kb >> 6) & 1;

    // Issue async loads for the next chunk into the alternate buffer.
    if (kb + 64 < nkeys_blk) {
      stage_kv(ldsK + (cur ^ 1) * 8192u, ldsV + (cur ^ 1) * 8192u, kbase_g,
               vbase_g, kb + 64, w, srow, scol);
    }

    if (kb < q0w + 16) {  // this wave still has visible keys in the chunk
      const unsigned short* Kc = &Klds[cur][0];
      const unsigned short* Vc = &Vlds[cur][0];

      // ---- scores S = Q @ K^T (scale folded into Q), batched loads ----
      v8f s[4];
      Frag kf[4];
#pragma unroll
      for (int t = 0; t < 4; ++t) {
        const unsigned short* krow = Kc + (16 * t + lnm) * 64 + kbase8;
        kf[t].q[0] = *(const uint4*)(krow);
        kf[t].q[1] = *(const uint4*)(krow + 16);
      }
#pragma unroll
      for (int t = 0; t < 4; ++t) s[t] = wmma_bf16(qa0.bf, kf[t].bf, v8f{});
#pragma unroll
      for (int t = 0; t < 4; ++t) {
        const unsigned short* krow = Kc + (16 * t + lnm) * 64 + 32 + kbase8;
        kf[t].q[0] = *(const uint4*)(krow);
        kf[t].q[1] = *(const uint4*)(krow + 16);
      }
#pragma unroll
      for (int t = 0; t < 4; ++t) s[t] = wmma_bf16(qa1.bf, kf[t].bf, s[t]);

      // ---- causal mask (IGNORE = -1e5 like the reference) ----
      if (kb + 63 > q0w) {
#pragma unroll
        for (int t = 0; t < 4; ++t) {
          const int key = kb + 16 * t + lnm;
#pragma unroll
          for (int r = 0; r < 8; ++r) {
            const int q = q0w + r + 8 * lhi;  // C layout: M = r + 8*hi
            if (key > q) s[t][r] = -1e5f;
          }
        }
      }

      // ---- chunk row max (reduce across the 16-lane half) ----
      float cm[8];
#pragma unroll
      for (int r = 0; r < 8; ++r) {
        float v = fmaxf(fmaxf(s[0][r], s[1][r]), fmaxf(s[2][r], s[3][r]));
        v = fmaxf(v, __shfl_xor(v, 1, 32));
        v = fmaxf(v, __shfl_xor(v, 2, 32));
        v = fmaxf(v, __shfl_xor(v, 4, 32));
        v = fmaxf(v, __shfl_xor(v, 8, 32));
        cm[r] = v;
      }
      float corr[8];
#pragma unroll
      for (int r = 0; r < 8; ++r) {
        const float mn = fmaxf(mrun[r], cm[r]);
        corr[r] = __expf(mrun[r] - mn);
        mrun[r] = mn;
      }

      // ---- P = exp(S - m), row sums, LDS C->A relayout ----
      float rs[8];
#pragma unroll
      for (int r = 0; r < 8; ++r) rs[r] = 0.0f;
#pragma unroll
      for (int t = 0; t < 4; ++t) {
#pragma unroll
        for (int r = 0; r < 8; ++r) {
          const float p = __expf(s[t][r] - mrun[r]);
          rs[r] += p;
          P[(r + 8 * lhi) * 64 + 16 * t + lnm] = f2bf(p);
        }
      }
#pragma unroll
      for (int r = 0; r < 8; ++r) {
        float v = rs[r];
        v += __shfl_xor(v, 1, 32);
        v += __shfl_xor(v, 2, 32);
        v += __shfl_xor(v, 4, 32);
        v += __shfl_xor(v, 8, 32);
        lrun[r] = lrun[r] * corr[r] + v;
      }

      // ---- rescale running O ----
#pragma unroll
      for (int t = 0; t < 4; ++t)
#pragma unroll
        for (int r = 0; r < 8; ++r) acc[t][r] *= corr[r];

      // ---- O += P @ V (batched B loads per kc pass) ----
      asm volatile("s_wait_dscnt 0x0" ::: "memory");  // P tile RAW (wave-priv)
#pragma unroll
      for (int kc = 0; kc < 64; kc += 32) {
        Frag pa;
        const unsigned short* prow = P + lnm * 64 + kc + kbase8;
        pa.q[0] = *(const uint4*)(prow);
        pa.q[1] = *(const uint4*)(prow + 16);
        Frag vb[4];
#pragma unroll
        for (int t = 0; t < 4; ++t) {
          const unsigned short* vrow = Vc + (16 * t + lnm) * 64 + kc + kbase8;
          vb[t].q[0] = *(const uint4*)(vrow);
          vb[t].q[1] = *(const uint4*)(vrow + 16);
        }
#pragma unroll
        for (int t = 0; t < 4; ++t)
          acc[t] = wmma_bf16(pa.bf, vb[t].bf, acc[t]);
      }
    }

    // Next chunk landed + everyone done reading current buffers.
    asm volatile("s_wait_asynccnt 0x0" ::: "memory");
    __syncthreads();
  }

  // ---- normalize and store Z[b, q, h*64 + d] as bf16 (A-major for kernel 3)
#pragma unroll
  for (int r = 0; r < 8; ++r) lrun[r] = 1.0f / lrun[r];
#pragma unroll
  for (int t = 0; t < 4; ++t) {
#pragma unroll
    for (int r = 0; r < 8; ++r) {
      const int q = q0w + r + 8 * lhi;
      Z[(size_t)(b * SEQ_LEN + q) * D_MODEL_C + h * D_HEAD_C + 16 * t + lnm] =
          f2bf(acc[t][r] * lrun[r]);
    }
  }
}

// ---------------------------------------------------------------------------
// Kernel 3: output projection.  Z (B*S x 768, bf16) @ W_O (768 x 768) + b_O.
// ---------------------------------------------------------------------------
__global__ __launch_bounds__(256) void proj_kernel(
    const unsigned short* __restrict__ Z, const float* __restrict__ Wo,
    const float* __restrict__ bO, float* __restrict__ out) {
  __shared__ unsigned short Wlds[D_HEAD_C * 32];  // [n=64][k=32] bf16, N-major

  const int tid  = threadIdx.x;
  const int w    = tid >> 5;
  const int lane = tid & 31;
  const int lnm  = lane & 15;
  const int lhi  = lane >> 4;
  const int row0 = blockIdx.x * 128 + 16 * w;
  const int n0   = blockIdx.y * 64;
  const int kbase8  = lhi ? 8 : 0;
  const int kbase16 = lhi ? 16 : 0;
  const unsigned short* zrow = Z + (size_t)(row0 + lnm) * D_MODEL_C;

  v8f acc[4] = {v8f{}, v8f{}, v8f{}, v8f{}};

  for (int k0 = 0; k0 < D_MODEL_C; k0 += 32) {
    {
      const int ml = tid & 31;
      const int d0 = (tid >> 5) * 8;
      const float* wr = Wo + (size_t)(k0 + ml) * D_MODEL_C + n0 + d0;
      float4 w0 = *(const float4*)wr;
      float4 w1 = *(const float4*)(wr + 4);
      Wlds[(d0 + 0) * 32 + ml] = f2bf(w0.x);
      Wlds[(d0 + 1) * 32 + ml] = f2bf(w0.y);
      Wlds[(d0 + 2) * 32 + ml] = f2bf(w0.z);
      Wlds[(d0 + 3) * 32 + ml] = f2bf(w0.w);
      Wlds[(d0 + 4) * 32 + ml] = f2bf(w1.x);
      Wlds[(d0 + 5) * 32 + ml] = f2bf(w1.y);
      Wlds[(d0 + 6) * 32 + ml] = f2bf(w1.z);
      Wlds[(d0 + 7) * 32 + ml] = f2bf(w1.w);
    }
    __syncthreads();

    Frag a;
    a.q[0] = *(const uint4*)(zrow + k0 + kbase8);
    a.q[1] = *(const uint4*)(zrow + k0 + kbase8 + 16);

    Frag bf[4];
#pragma unroll
    for (int t = 0; t < 4; ++t) {
      const unsigned short* brow = Wlds + (16 * t + lnm) * 32 + kbase8;
      bf[t].q[0] = *(const uint4*)(brow);
      bf[t].q[1] = *(const uint4*)(brow + 16);
    }
#pragma unroll
    for (int t = 0; t < 4; ++t) acc[t] = wmma_bf16(a.bf, bf[t].bf, acc[t]);
    __syncthreads();
  }

#pragma unroll
  for (int t = 0; t < 4; ++t) {
    const float bv = bO[n0 + 16 * t + lnm];
#pragma unroll
    for (int r = 0; r < 8; ++r) {
      out[(size_t)(row0 + r + 8 * lhi) * D_MODEL_C + n0 + 16 * t + lnm] =
          acc[t][r] + bv;
    }
  }
}

// ---------------------------------------------------------------------------
extern "C" void kernel_launch(void* const* d_in, const int* in_sizes, int n_in,
                              void* d_out, int out_size, void* d_ws,
                              size_t ws_size, hipStream_t stream) {
  (void)in_sizes; (void)n_in; (void)out_size; (void)ws_size;
  const float* x  = (const float*)d_in[0];
  const float* WQ = (const float*)d_in[1];
  const float* WK = (const float*)d_in[2];
  const float* WV = (const float*)d_in[3];
  const float* WO = (const float*)d_in[4];
  const float* bQ = (const float*)d_in[5];
  const float* bK = (const float*)d_in[6];
  const float* bV = (const float*)d_in[7];
  const float* bO = (const float*)d_in[8];
  float* out = (float*)d_out;

  const size_t per = (size_t)BATCH_C * N_HEADS_C * SEQ_LEN * D_HEAD_C;  // 6.29M
  unsigned short* Qb = (unsigned short*)d_ws;          // [b,h,s,d] bf16 (scaled)
  unsigned short* Kb = Qb + per;                        // [b,h,s,d] bf16
  unsigned short* Vt = Kb + per;                        // [b,h,d,s] bf16
  unsigned short* Z  = Vt + per;                        // [b,s,768] bf16

  dim3 blk(256);
  dim3 g1(BATCH_C * (SEQ_LEN / 128), N_HEADS_C);
  qkv_kernel<<<g1, blk, 0, stream>>>(x, WQ, bQ, Qb, 0);
  qkv_kernel<<<g1, blk, 0, stream>>>(x, WK, bK, Kb, 1);
  qkv_kernel<<<g1, blk, 0, stream>>>(x, WV, bV, Vt, 2);

  dim3 g2(SEQ_LEN / 128, N_HEADS_C, BATCH_C);
  attn_kernel<<<g2, blk, 0, stream>>>(Qb, Kb, Vt, Z);

  dim3 g3(BATCH_C * SEQ_LEN / 128, D_MODEL_C / 64);
  proj_kernel<<<g3, blk, 0, stream>>>(Z, WO, bO, out);
}
